// FrameAwareNodeSelectionGNN_48696339202116
// MI455X (gfx1250) — matrix operand order repaired and checked by reading, not compiled, not run on hardware
//
#include <hip/hip_runtime.h>

#define NN 100000
#define DD 64
#define EE 800000
#define EDD 32
#define LL 5
#define LN_EPS 1e-5f

typedef __attribute__((ext_vector_type(16))) __bf16 v16bf;
typedef __attribute__((ext_vector_type(8)))  float  v8f;

static __device__ __forceinline__ unsigned short f2bf(float f) {
  unsigned int u = __builtin_bit_cast(unsigned int, f);
  unsigned int r = u + 0x7FFFu + ((u >> 16) & 1u);   // round-to-nearest-even
  return (unsigned short)(r >> 16);
}

union FragAB { v16bf v; unsigned int u32[8]; };

// A fragment (16-bit A, 16xK tile, chunk k0..k0+31) per ISA 7.12.2:
// lane = half*16 + m ; VGPR v<4: K = 2v+s+8h ; v>=4: K = 16+2(v-4)+s+8h
static __device__ __forceinline__ v16bf loadA(const unsigned short* smA, int strideA,
                                              int k0, int lane) {
  int m = lane & 15, h = lane >> 4;
  FragAB f;
#pragma unroll
  for (int v = 0; v < 8; ++v) {
    int kb = (v < 4) ? (2 * v + 8 * h) : (16 + 2 * (v - 4) + 8 * h);
    f.u32[v] = *(const unsigned int*)(smA + m * strideA + k0 + kb); // 2 bf16, even addr
  }
  return f.v;
}

// B fragment from TRANSPOSED weights smBt[n][k] (bf16). 32x16 B tile, cols n0..n0+15,
// chunk k0: lane = half*16 + n ; VGPR v<4: K = k0 + 2v+s+8h ; v>=4: K = k0 + 16+2(v-4)+s+8h
static __device__ __forceinline__ v16bf loadB(const unsigned short* smBt, int strideB,
                                              int n0, int k0, int lane) {
  int n = lane & 15, h = lane >> 4;
  FragAB f;
#pragma unroll
  for (int v = 0; v < 8; ++v) {
    int k = k0 + ((v < 4) ? (2 * v + 8 * h) : (16 + 2 * (v - 4) + 8 * h));
    f.u32[v] = *(const unsigned int*)(smBt + (n0 + n) * strideB + k);
  }
  return f.v;
}

static __device__ __forceinline__ v8f wmma_bf16(v16bf a, v16bf b, v8f c) {
  return __builtin_amdgcn_wmma_f32_16x16x32_bf16(false, a, false, b, (short)0, c,
                                                 false, false);
}

// ---------------- Kernel 1: h = leaky(x@W1+b1)@W2 + b2 ----------------
__global__ __launch_bounds__(128)
void node_mlp_kernel(const float* __restrict__ x, const float* __restrict__ W1,
                     const float* __restrict__ b1, const float* __restrict__ W2,
                     const float* __restrict__ b2, float* __restrict__ h_out) {
  __shared__ unsigned short sW1t[128 * 64];      // [n][k], 16 KB
  __shared__ unsigned short sW2t[64 * 128];      // [n][k], 16 KB
  __shared__ float sb1[128];
  __shared__ float sb2[64];
  __shared__ unsigned short sX[4][16 * 64];      // per-wave x tile (bf16), 8 KB
  __shared__ unsigned short sMid[4][16 * 128];   // per-wave mid tile (bf16), 16 KB

  int tid = threadIdx.x;
  for (int i = tid; i < 64 * 128; i += 128) {    // W1[k][n] -> sW1t[n*64+k]
    int k = i >> 7, n = i & 127;
    sW1t[n * 64 + k] = f2bf(W1[i]);
  }
  for (int i = tid; i < 128 * 64; i += 128) {    // W2[k][n] -> sW2t[n*128+k]
    int k = i >> 6, n = i & 63;
    sW2t[n * 128 + k] = f2bf(W2[i]);
  }
  if (tid < 128) sb1[tid] = b1[tid];
  if (tid < 64)  sb2[tid] = b2[tid];
  __syncthreads();

  int wave = tid >> 5, lane = tid & 31;
  int ncol = lane & 15, half = lane >> 4;
  int row0 = blockIdx.x * 64 + wave * 16;

  unsigned short* myX = sX[wave];
  for (int i = lane; i < 16 * 64; i += 32) {
    int r = i >> 6, c = i & 63;
    int gr = row0 + r;
    myX[i] = f2bf((gr < NN) ? x[(size_t)gr * 64 + c] : 0.f);
  }
  __syncthreads();

  v16bf a0 = loadA(myX, 64, 0, lane);
  v16bf a1 = loadA(myX, 64, 32, lane);
  unsigned short* myM = sMid[wave];
#pragma unroll
  for (int nt = 0; nt < 8; ++nt) {               // mid cols 0..127
    v8f acc = {};
    acc = wmma_bf16(a0, loadB(sW1t, 64, nt * 16, 0, lane), acc);
    acc = wmma_bf16(a1, loadB(sW1t, 64, nt * 16, 32, lane), acc);
    int j = nt * 16 + ncol;
    float bb = sb1[j];
#pragma unroll
    for (int v = 0; v < 8; ++v) {
      float t = acc[v] + bb;
      t = (t < 0.f) ? 0.1f * t : t;
      myM[(v + 8 * half) * 128 + j] = f2bf(t);
    }
  }
  __syncthreads();

  v16bf ma0 = loadA(myM, 128, 0, lane);
  v16bf ma1 = loadA(myM, 128, 32, lane);
  v16bf ma2 = loadA(myM, 128, 64, lane);
  v16bf ma3 = loadA(myM, 128, 96, lane);
#pragma unroll
  for (int nt = 0; nt < 4; ++nt) {               // h cols 0..63
    v8f acc = {};
    acc = wmma_bf16(ma0, loadB(sW2t, 128, nt * 16, 0, lane), acc);
    acc = wmma_bf16(ma1, loadB(sW2t, 128, nt * 16, 32, lane), acc);
    acc = wmma_bf16(ma2, loadB(sW2t, 128, nt * 16, 64, lane), acc);
    acc = wmma_bf16(ma3, loadB(sW2t, 128, nt * 16, 96, lane), acc);
    int j = nt * 16 + ncol;
    float bb = sb2[j];
#pragma unroll
    for (int v = 0; v < 8; ++v) {
      int gr = row0 + v + 8 * half;
      if (gr < NN) h_out[(size_t)gr * 64 + j] = acc[v] + bb;
    }
  }
}

// ------- Kernel 2: edge MLP (WMMA) fused with gather/scale/scatter-add -------
__global__ __launch_bounds__(256)
void edge_kernel(const float* __restrict__ edge_attr, const long long* __restrict__ ei,
                 const float* __restrict__ We1, const float* __restrict__ be1,
                 const float* __restrict__ We2, const float* __restrict__ be2,
                 const float* __restrict__ dscale_arr, int layer,
                 const float* __restrict__ h, float* __restrict__ agg) {
  __shared__ unsigned short sWe1t[64 * 32];      // [n][k], 4 KB
  __shared__ unsigned short sWe2t[64 * 64];      // [n][k], 8 KB
  __shared__ float sbe1[64], sbe2[64];
  __shared__ unsigned short sAttr[8][16 * 32];   // per-wave attr tile, 8 KB
  __shared__ unsigned short sT[8][16 * 64];      // per-wave mid tile, 16 KB
  __shared__ int sSrc[8][16], sDst[8][16];
  __shared__ float sScale;

  int tid = threadIdx.x;
  for (int i = tid; i < 32 * 64; i += 256) {     // We1[k][n] -> sWe1t[n*32+k]
    int k = i >> 6, n = i & 63;
    sWe1t[n * 32 + k] = f2bf(We1[i]);
  }
  for (int i = tid; i < 64 * 64; i += 256) {     // We2[k][n] -> sWe2t[n*64+k]
    int k = i >> 6, n = i & 63;
    sWe2t[n * 64 + k] = f2bf(We2[i]);
  }
  if (tid < 64) { sbe1[tid] = be1[tid]; sbe2[tid] = be2[tid]; }
  if (tid == 0) sScale = dscale_arr[layer];

  int wave = tid >> 5, lane = tid & 31;
  int ncol = lane & 15, half = lane >> 4;
  long long e0 = (long long)blockIdx.x * 128 + wave * 16;

  unsigned short* myA = sAttr[wave];
  for (int i = lane; i < 16 * 32; i += 32) {
    int r = i >> 5, c = i & 31;
    long long e = e0 + r;
    myA[i] = f2bf((e < EE) ? edge_attr[e * 32 + c] : 0.f);
  }
  if (lane < 16) {
    long long e = e0 + lane;
    if (e < EE) {
      sSrc[wave][lane] = (int)ei[e];             // edge_index[0][e]
      sDst[wave][lane] = (int)ei[(long long)EE + e];
    } else { sSrc[wave][lane] = 0; sDst[wave][lane] = 0; }
  }
  __syncthreads();

  v16bf a = loadA(myA, 32, 0, lane);
  unsigned short* myT = sT[wave];
#pragma unroll
  for (int nt = 0; nt < 4; ++nt) {               // t = leaky(attr@We1+be1)
    v8f acc = {};
    acc = wmma_bf16(a, loadB(sWe1t, 32, nt * 16, 0, lane), acc);
    int j = nt * 16 + ncol;
    float bb = sbe1[j];
#pragma unroll
    for (int v = 0; v < 8; ++v) {
      float t = acc[v] + bb;
      t = (t < 0.f) ? 0.1f * t : t;
      myT[(v + 8 * half) * 64 + j] = f2bf(t);
    }
  }
  __syncthreads();

  v16bf t0 = loadA(myT, 64, 0, lane);
  v16bf t1 = loadA(myT, 64, 32, lane);
  float scale = sScale;
#pragma unroll
  for (int nt = 0; nt < 4; ++nt) {               // e_feat = t@We2+be2 ; fuse msg/scatter
    v8f acc = {};
    acc = wmma_bf16(t0, loadB(sWe2t, 64, nt * 16, 0, lane), acc);
    acc = wmma_bf16(t1, loadB(sWe2t, 64, nt * 16, 32, lane), acc);
    int j = nt * 16 + ncol;
    float bb = sbe2[j];
#pragma unroll
    for (int v = 0; v < 8; ++v) {
      int m = v + 8 * half;
      long long e = e0 + m;
      if (e < EE) {
        int s = sSrc[wave][m], d = sDst[wave][m];
        float msg = (acc[v] + bb) * scale * h[(size_t)s * 64 + j];
        atomicAdd(&agg[(size_t)d * 64 + j], msg);
      }
    }
  }
}

// ---------------- Kernel 3: residual + LayerNorm (1 wave per row) ----------------
__global__ __launch_bounds__(256)
void norm_kernel(const float* __restrict__ x_in, const float* __restrict__ agg,
                 const float* __restrict__ rw_arr, int layer,
                 const float* __restrict__ ln_g, const float* __restrict__ ln_b,
                 float* __restrict__ x_out) {
  int wave = threadIdx.x >> 5, lane = threadIdx.x & 31;
  long long row = (long long)blockIdx.x * 8 + wave;
  if (row >= NN) return;
  float rw = rw_arr[layer];
  size_t base = (size_t)row * 64;
  float v0 = x_in[base + lane]      + agg[base + lane]      * rw;
  float v1 = x_in[base + lane + 32] + agg[base + lane + 32] * rw;
  float s = v0 + v1;
#pragma unroll
  for (int off = 16; off; off >>= 1) s += __shfl_down(s, off, 32);
  float mu = __shfl(s, 0, 32) * (1.f / 64.f);
  float d0 = v0 - mu, d1 = v1 - mu;
  float q = d0 * d0 + d1 * d1;
#pragma unroll
  for (int off = 16; off; off >>= 1) q += __shfl_down(q, off, 32);
  float rs = rsqrtf(__shfl(q, 0, 32) * (1.f / 64.f) + LN_EPS);
  x_out[base + lane]      = d0 * rs * ln_g[lane]      + ln_b[lane];
  x_out[base + lane + 32] = d1 * rs * ln_g[lane + 32] + ln_b[lane + 32];
}

extern "C" void kernel_launch(void* const* d_in, const int* in_sizes, int n_in,
                              void* d_out, int out_size, void* d_ws, size_t ws_size,
                              hipStream_t stream) {
  (void)in_sizes; (void)n_in; (void)out_size; (void)ws_size;
  const float*      x      = (const float*)d_in[0];
  const long long*  ei     = (const long long*)d_in[1];
  const float*      ea     = (const float*)d_in[2];
  const float*      W1     = (const float*)d_in[3];
  const float*      b1     = (const float*)d_in[4];
  const float*      W2     = (const float*)d_in[5];
  const float*      b2     = (const float*)d_in[6];
  const float*      We1    = (const float*)d_in[7];
  const float*      be1    = (const float*)d_in[8];
  const float*      We2    = (const float*)d_in[9];
  const float*      be2    = (const float*)d_in[10];
  const float*      res_w  = (const float*)d_in[11];
  const float*      dscale = (const float*)d_in[12];
  const float*      ln_g   = (const float*)d_in[13];
  const float*      ln_b   = (const float*)d_in[14];
  float* xout = (float*)d_out;

  float* h   = (float*)d_ws;                    // N*64 f32
  float* agg = h + (size_t)NN * 64;             // N*64 f32

  for (int l = 0; l < LL; ++l) {
    const float* xin = (l == 0) ? x : xout;
    hipMemsetAsync(agg, 0, (size_t)NN * 64 * sizeof(float), stream);
    node_mlp_kernel<<<(NN + 63) / 64, 128, 0, stream>>>(
        xin, W1 + (size_t)l * 64 * 128, b1 + (size_t)l * 128,
        W2 + (size_t)l * 128 * 64, b2 + (size_t)l * 64, h);
    edge_kernel<<<EE / 128, 256, 0, stream>>>(
        ea, ei, We1 + (size_t)l * 32 * 64, be1 + (size_t)l * 64,
        We2 + (size_t)l * 64 * 64, be2 + (size_t)l * 64, dscale, l, h, agg);
    norm_kernel<<<(NN + 7) / 8, 256, 0, stream>>>(
        xin, agg, res_w, l, ln_g + (size_t)l * 64, ln_b + (size_t)l * 64, xout);
  }
}
